// GATLinkPredictor_34634616275043
// MI455X (gfx1250) — matrix operand order repaired and checked by reading, not compiled, not run on hardware
//
#include <hip/hip_runtime.h>
#include <cfloat>

typedef __attribute__((ext_vector_type(16))) __bf16 v16bf;
typedef __attribute__((ext_vector_type(8)))  float  v8f;

#define CDIV(a,b) (((a)+(b)-1)/(b))

static __device__ __forceinline__ unsigned short f2bf(float f) {
  unsigned int u = __float_as_uint(f);
  unsigned int r = (u + 0x7FFFu + ((u >> 16) & 1u)) >> 16;  // RNE
  return (unsigned short)r;
}

static __device__ __forceinline__ void atomicMaxF(float* addr, float val) {
  unsigned int* a = (unsigned int*)addr;
  unsigned int old = __float_as_uint(*addr);
  while (__uint_as_float(old) < val) {
    unsigned int assumed = old;
    old = atomicCAS(a, assumed, __float_as_uint(val));
    if (old == assumed) break;
  }
}

// ---------------- elementwise conversions / init ----------------

__global__ void k_f32_to_bf16(const float* __restrict__ in,
                              unsigned short* __restrict__ out, int n) {
  int i = blockIdx.x * blockDim.x + threadIdx.x;
  if (i < n) out[i] = f2bf(in[i]);
}

__global__ void k_fill_f32(float* __restrict__ p, float v, int n) {
  int i = blockIdx.x * blockDim.x + threadIdx.x;
  if (i < n) p[i] = v;
}

// Pack W [K, Nout] (f32, row-major) into the bf16 B-operand order used here (contiguous k, matches A):
// pk[((nt*KC + kc)*32 + lane)*16 + j] = W[(kc*32 + (lane>>4)*16 + j)*Nout + nt*16 + (lane&15)]
__global__ void k_pack_w_bf16(const float* __restrict__ W,
                              unsigned short* __restrict__ pk, int K, int Nout) {
  int total = (Nout / 16) * (K / 32) * 32 * 16;
  int i = blockIdx.x * blockDim.x + threadIdx.x;
  if (i >= total) return;
  int j    = i & 15;
  int lane = (i >> 4) & 31;
  int rest = i >> 9;
  int KC   = K >> 5;
  int kc   = rest % KC;
  int nt   = rest / KC;
  int k = kc * 32 + (lane >> 4) * 16 + j;
  int n = nt * 16 + (lane & 15);
  pk[i] = f2bf(W[k * Nout + n]);
}

// ---------------- WMMA GEMM: out[M, NT*16] = A[M, KC*32](bf16) * Bpk + bias ----------
// One wave computes a 16-row strip across all NT column tiles. KC compile-time so all
// global loads use immediate instruction offsets from a single per-lane base address.
template <int NT, int KC>
__global__ void k_wmma_gemm_bf16(const unsigned short* __restrict__ A,
                                 const unsigned short* __restrict__ Bpk,
                                 const float* __restrict__ bias,
                                 float* __restrict__ out, int M) {
  constexpr int K    = KC * 32;
  constexpr int Nout = NT * 16;
  const int lane = threadIdx.x & 31;
  const int wave = threadIdx.x >> 5;
  const int tile = blockIdx.x * (blockDim.x >> 5) + wave;
  if (tile * 16 >= M) return;  // whole-wave exit: EXEC stays all-1s for WMMA
  const int row  = tile * 16 + (lane & 15);
  const int koff = (lane >> 4) * 16;  // k order used here, same as Bpk: lanes<16 K{0..15}, lanes>=16 K{16..31}

  const unsigned short* aBase = A + (size_t)row * K + koff;
  const unsigned short* bBase = Bpk + lane * 16;

  v8f acc[NT];
#pragma unroll
  for (int t = 0; t < NT; ++t)
#pragma unroll
    for (int r = 0; r < 8; ++r) acc[t][r] = 0.0f;

#pragma unroll
  for (int kc = 0; kc < KC; ++kc) {
    union { v16bf v; uint4 q[2]; } a;
    a.q[0] = *(const uint4*)(aBase + kc * 32);       // imm offset kc*64 B
    a.q[1] = *(const uint4*)(aBase + kc * 32 + 8);  // imm offset kc*64+16 B
#pragma unroll
    for (int t = 0; t < NT; ++t) {
      union { v16bf v; uint4 q[2]; } b;
      const unsigned short* bp = bBase + (t * KC + kc) * 512;  // imm offset (t*KC+kc)*1024 B
      b.q[0] = *(const uint4*)(bp);
      b.q[1] = *(const uint4*)(bp + 8);
      acc[t] = __builtin_amdgcn_wmma_f32_16x16x32_bf16(
          false, a.v, false, b.v, (short)0, acc[t], false, false);
    }
  }

  // C/D layout: N = lane&15; VGPR r holds M=r (lanes 0-15) / M=8+r (lanes 16-31)
  const int colBase = lane & 15;
  const int rowHalf = (lane >> 4) * 8;
#pragma unroll
  for (int t = 0; t < NT; ++t) {
    int col  = t * 16 + colBase;
    float bb = bias[col];
    float* oBase = out + (size_t)(tile * 16 + rowHalf) * Nout + col;
#pragma unroll
    for (int r = 0; r < 8; ++r) {
      oBase[r * Nout] = acc[t][r] + bb;  // imm offset r*Nout*4 B
    }
  }
}

// ---------------- Layer-1 edge passes (4 heads x 32 ch) ----------------
// wave per edge; lane covers 4 consecutive channels; head = lane>>3

__global__ void k_edge_logits_l1(const long long* __restrict__ ei, int E, int Etot,
                                 const float* __restrict__ xl, const float* __restrict__ xr,
                                 const float* __restrict__ att,
                                 float* __restrict__ logits, float* __restrict__ mx) {
  int lane = threadIdx.x & 31;
  int e = (blockIdx.x * blockDim.x + threadIdx.x) >> 5;
  if (e >= Etot) return;
  int src, dst;
  if (e < E) { src = (int)ei[e]; dst = (int)ei[E + e]; } else { src = dst = e - E; }
  int c = lane * 4;
  float4 a = *(const float4*)(xl + (size_t)src * 128 + c);
  float4 b = *(const float4*)(xr + (size_t)dst * 128 + c);
  int h  = lane >> 3;
  int cw = (lane & 7) * 4;
  float4 at = *(const float4*)(att + h * 32 + cw);
  float p = 0.0f, v;
  v = a.x + b.x; v = v > 0.0f ? v : 0.2f * v; p += v * at.x;
  v = a.y + b.y; v = v > 0.0f ? v : 0.2f * v; p += v * at.y;
  v = a.z + b.z; v = v > 0.0f ? v : 0.2f * v; p += v * at.z;
  v = a.w + b.w; v = v > 0.0f ? v : 0.2f * v; p += v * at.w;
  p += __shfl_xor(p, 1);
  p += __shfl_xor(p, 2);
  p += __shfl_xor(p, 4);
  if ((lane & 7) == 0) {
    logits[(size_t)e * 4 + h] = p;
    atomicMaxF(mx + (size_t)dst * 4 + h, p);
  }
}

__global__ void k_edge_softmax_l1(const long long* __restrict__ ei, int E, int Etot,
                                  float* __restrict__ logits, const float* __restrict__ mx,
                                  float* __restrict__ denom) {
  int i = blockIdx.x * blockDim.x + threadIdx.x;
  if (i >= Etot * 4) return;
  int e = i >> 2, h = i & 3;
  int dst = (e < E) ? (int)ei[E + e] : (e - E);
  float ex = __expf(logits[i] - mx[(size_t)dst * 4 + h]);
  logits[i] = ex;  // reuse buffer for exp values
  atomicAdd(denom + (size_t)dst * 4 + h, ex);
}

__global__ void k_edge_aggregate_l1(const long long* __restrict__ ei, int E, int Etot,
                                    const float* __restrict__ xl, const float* __restrict__ ex,
                                    float* __restrict__ agg) {
  int lane = threadIdx.x & 31;
  int e = (blockIdx.x * blockDim.x + threadIdx.x) >> 5;
  if (e >= Etot) return;
  int src, dst;
  if (e < E) { src = (int)ei[e]; dst = (int)ei[E + e]; } else { src = dst = e - E; }
  int h = lane >> 3;
  float w = ex[(size_t)e * 4 + h];
  int c = lane * 4;
  float4 a = *(const float4*)(xl + (size_t)src * 128 + c);
  float* o = agg + (size_t)dst * 128 + c;
  atomicAdd(o + 0, a.x * w);
  atomicAdd(o + 1, a.y * w);
  atomicAdd(o + 2, a.z * w);
  atomicAdd(o + 3, a.w * w);
}

// finalize: h = elu(layernorm(agg/denom + bias)); store bf16 for layer-2 GEMM
__global__ void k_node_finalize_l1(const float* __restrict__ agg, const float* __restrict__ denom,
                                   const float* __restrict__ bias, const float* __restrict__ g,
                                   const float* __restrict__ be,
                                   unsigned short* __restrict__ hbf, int N) {
  int lane = threadIdx.x & 31;
  int n = (blockIdx.x * blockDim.x + threadIdx.x) >> 5;
  if (n >= N) return;
  int c = lane * 4;
  int h = lane >> 3;
  float d = denom[(size_t)n * 4 + h] + 1e-16f;
  float4 a = *(const float4*)(agg + (size_t)n * 128 + c);
  float v[4];
  v[0] = a.x / d + bias[c + 0];
  v[1] = a.y / d + bias[c + 1];
  v[2] = a.z / d + bias[c + 2];
  v[3] = a.w / d + bias[c + 3];
  float s = v[0] + v[1] + v[2] + v[3];
#pragma unroll
  for (int m = 1; m < 32; m <<= 1) s += __shfl_xor(s, m);
  float mu = s * (1.0f / 128.0f);
  float q = 0.0f;
#pragma unroll
  for (int j = 0; j < 4; ++j) { float t = v[j] - mu; q += t * t; }
#pragma unroll
  for (int m = 1; m < 32; m <<= 1) q += __shfl_xor(q, m);
  float rstd = __frsqrt_rn(q * (1.0f / 128.0f) + 1e-5f);
#pragma unroll
  for (int j = 0; j < 4; ++j) {
    float y = (v[j] - mu) * rstd * g[c + j] + be[c + j];
    y = y > 0.0f ? y : (__expf(y) - 1.0f);  // ELU
    hbf[(size_t)n * 128 + c + j] = f2bf(y);
  }
}

// ---------------- Layer-2 edge passes (1 head x 64 ch) ----------------
// wave per edge; lane covers 2 consecutive channels

__global__ void k_edge_logits_l2(const long long* __restrict__ ei, int E, int Etot,
                                 const float* __restrict__ hl, const float* __restrict__ hr,
                                 const float* __restrict__ att,
                                 float* __restrict__ logits, float* __restrict__ mx) {
  int lane = threadIdx.x & 31;
  int e = (blockIdx.x * blockDim.x + threadIdx.x) >> 5;
  if (e >= Etot) return;
  int src, dst;
  if (e < E) { src = (int)ei[e]; dst = (int)ei[E + e]; } else { src = dst = e - E; }
  int c = lane * 2;
  float2 a  = *(const float2*)(hl + (size_t)src * 64 + c);
  float2 b  = *(const float2*)(hr + (size_t)dst * 64 + c);
  float2 at = *(const float2*)(att + c);
  float p = 0.0f, v;
  v = a.x + b.x; v = v > 0.0f ? v : 0.2f * v; p += v * at.x;
  v = a.y + b.y; v = v > 0.0f ? v : 0.2f * v; p += v * at.y;
#pragma unroll
  for (int m = 1; m < 32; m <<= 1) p += __shfl_xor(p, m);
  if (lane == 0) {
    logits[e] = p;
    atomicMaxF(mx + dst, p);
  }
}

__global__ void k_edge_softmax_l2(const long long* __restrict__ ei, int E, int Etot,
                                  float* __restrict__ logits, const float* __restrict__ mx,
                                  float* __restrict__ denom) {
  int e = blockIdx.x * blockDim.x + threadIdx.x;
  if (e >= Etot) return;
  int dst = (e < E) ? (int)ei[E + e] : (e - E);
  float ex = __expf(logits[e] - mx[dst]);
  logits[e] = ex;
  atomicAdd(denom + dst, ex);
}

__global__ void k_edge_aggregate_l2(const long long* __restrict__ ei, int E, int Etot,
                                    const float* __restrict__ hl, const float* __restrict__ ex,
                                    float* __restrict__ agg) {
  int lane = threadIdx.x & 31;
  int e = (blockIdx.x * blockDim.x + threadIdx.x) >> 5;
  if (e >= Etot) return;
  int src, dst;
  if (e < E) { src = (int)ei[e]; dst = (int)ei[E + e]; } else { src = dst = e - E; }
  float w = ex[e];
  int c = lane * 2;
  float2 a = *(const float2*)(hl + (size_t)src * 64 + c);
  float* o = agg + (size_t)dst * 64 + c;
  atomicAdd(o + 0, a.x * w);
  atomicAdd(o + 1, a.y * w);
}

// z = layernorm(agg/denom + bias2)
__global__ void k_node_finalize_l2(const float* __restrict__ agg, const float* __restrict__ denom,
                                   const float* __restrict__ bias, const float* __restrict__ g,
                                   const float* __restrict__ be,
                                   float* __restrict__ z, int N) {
  int lane = threadIdx.x & 31;
  int n = (blockIdx.x * blockDim.x + threadIdx.x) >> 5;
  if (n >= N) return;
  int c = lane * 2;
  float d = denom[n] + 1e-16f;
  float2 a = *(const float2*)(agg + (size_t)n * 64 + c);
  float v0 = a.x / d + bias[c + 0];
  float v1 = a.y / d + bias[c + 1];
  float s = v0 + v1;
#pragma unroll
  for (int m = 1; m < 32; m <<= 1) s += __shfl_xor(s, m);
  float mu = s * (1.0f / 64.0f);
  float t0 = v0 - mu, t1 = v1 - mu;
  float q = t0 * t0 + t1 * t1;
#pragma unroll
  for (int m = 1; m < 32; m <<= 1) q += __shfl_xor(q, m);
  float rstd = __frsqrt_rn(q * (1.0f / 64.0f) + 1e-5f);
  z[(size_t)n * 64 + c + 0] = t0 * rstd * g[c + 0] + be[c + 0];
  z[(size_t)n * 64 + c + 1] = t1 * rstd * g[c + 1] + be[c + 1];
}

// decode: out[i] = dot(z[eli0[i]], z[eli1[i]])   (wave per pair)
__global__ void k_decode(const long long* __restrict__ eli, int EL,
                         const float* __restrict__ z, float* __restrict__ out) {
  int lane = threadIdx.x & 31;
  int i = (blockIdx.x * blockDim.x + threadIdx.x) >> 5;
  if (i >= EL) return;
  int a = (int)eli[i];
  int b = (int)eli[EL + i];
  int c = lane * 2;
  float2 za = *(const float2*)(z + (size_t)a * 64 + c);
  float2 zb = *(const float2*)(z + (size_t)b * 64 + c);
  float p = za.x * zb.x + za.y * zb.y;
#pragma unroll
  for (int m = 1; m < 32; m <<= 1) p += __shfl_xor(p, m);
  if (lane == 0) out[i] = p;
}

// ---------------- host-side launcher ----------------

static inline size_t align256(size_t x) { return (x + 255) & ~(size_t)255; }

extern "C" void kernel_launch(void* const* d_in, const int* in_sizes, int n_in,
                              void* d_out, int out_size, void* d_ws, size_t ws_size,
                              hipStream_t stream) {
  const float*     x    = (const float*)d_in[0];
  const long long* ei   = (const long long*)d_in[1];
  const long long* eli  = (const long long*)d_in[2];
  const float* W1l  = (const float*)d_in[3];
  const float* b1l  = (const float*)d_in[4];
  const float* W1r  = (const float*)d_in[5];
  const float* b1r  = (const float*)d_in[6];
  const float* att1 = (const float*)d_in[7];
  const float* bias1= (const float*)d_in[8];
  const float* g1   = (const float*)d_in[9];
  const float* be1  = (const float*)d_in[10];
  const float* W2l  = (const float*)d_in[11];
  const float* b2l  = (const float*)d_in[12];
  const float* W2r  = (const float*)d_in[13];
  const float* b2r  = (const float*)d_in[14];
  const float* att2 = (const float*)d_in[15];
  const float* bias2= (const float*)d_in[16];
  const float* g2   = (const float*)d_in[17];
  const float* be2  = (const float*)d_in[18];
  float* out = (float*)d_out;

  const int N    = in_sizes[0] / 128;
  const int E    = in_sizes[1] / 2;
  const int EL   = in_sizes[2] / 2;
  const int Etot = E + N;

  // carve workspace
  char* p = (char*)d_ws;
  auto carve = [&](size_t bytes) { char* r = p; p += align256(bytes); return r; };
  unsigned short* x_bf    = (unsigned short*)carve((size_t)N * 128 * 2);
  unsigned short* h_bf    = (unsigned short*)carve((size_t)N * 128 * 2);
  unsigned short* w1l_pk  = (unsigned short*)carve(128 * 128 * 2);
  unsigned short* w1r_pk  = (unsigned short*)carve(128 * 128 * 2);
  unsigned short* w2l_pk  = (unsigned short*)carve(128 * 64 * 2);
  unsigned short* w2r_pk  = (unsigned short*)carve(128 * 64 * 2);
  float* xl1    = (float*)carve((size_t)N * 128 * 4);
  float* xr1    = (float*)carve((size_t)N * 128 * 4);
  float* hl2    = (float*)carve((size_t)N * 64 * 4);
  float* hr2    = (float*)carve((size_t)N * 64 * 4);
  float* lg1    = (float*)carve((size_t)Etot * 4 * 4);
  float* lg2    = (float*)carve((size_t)Etot * 4);
  float* m1     = (float*)carve((size_t)N * 4 * 4);
  float* dn1    = (float*)carve((size_t)N * 4 * 4);
  float* m2     = (float*)carve((size_t)N * 4);
  float* dn2    = (float*)carve((size_t)N * 4);
  float* agg1   = (float*)carve((size_t)N * 128 * 4);
  float* agg2   = (float*)carve((size_t)N * 64 * 4);
  float* zbuf   = (float*)carve((size_t)N * 64 * 4);
  (void)ws_size; (void)n_in;

  const int B = 256;

  // 1) convert x to bf16, pack weights
  k_f32_to_bf16<<<CDIV(N * 128, B), B, 0, stream>>>(x, x_bf, N * 128);
  k_pack_w_bf16<<<CDIV(128 * 128, B), B, 0, stream>>>(W1l, w1l_pk, 128, 128);
  k_pack_w_bf16<<<CDIV(128 * 128, B), B, 0, stream>>>(W1r, w1r_pk, 128, 128);
  k_pack_w_bf16<<<CDIV(128 * 64, B), B, 0, stream>>>(W2l, w2l_pk, 128, 64);
  k_pack_w_bf16<<<CDIV(128 * 64, B), B, 0, stream>>>(W2r, w2r_pk, 128, 64);

  // 2) layer-1 projections (WMMA): xl1 = x@W1l + b1l ; xr1 = x@W1r + b1r
  int tiles1 = CDIV(N, 16);
  k_wmma_gemm_bf16<8, 4><<<CDIV(tiles1, 8), B, 0, stream>>>(x_bf, w1l_pk, b1l, xl1, N);
  k_wmma_gemm_bf16<8, 4><<<CDIV(tiles1, 8), B, 0, stream>>>(x_bf, w1r_pk, b1r, xr1, N);

  // 3) layer-1 segment softmax + aggregation
  k_fill_f32<<<CDIV(N * 4, B), B, 0, stream>>>(m1, -FLT_MAX, N * 4);
  hipMemsetAsync(dn1, 0, (size_t)N * 4 * 4, stream);
  hipMemsetAsync(agg1, 0, (size_t)N * 128 * 4, stream);
  k_edge_logits_l1<<<CDIV(Etot * 32, B), B, 0, stream>>>(ei, E, Etot, xl1, xr1, att1, lg1, m1);
  k_edge_softmax_l1<<<CDIV(Etot * 4, B), B, 0, stream>>>(ei, E, Etot, lg1, m1, dn1);
  k_edge_aggregate_l1<<<CDIV(Etot * 32, B), B, 0, stream>>>(ei, E, Etot, xl1, lg1, agg1);
  k_node_finalize_l1<<<CDIV(N * 32, B), B, 0, stream>>>(agg1, dn1, bias1, g1, be1, h_bf, N);

  // 4) layer-2 projections (WMMA): hl2 = h@W2l + b2l ; hr2 = h@W2r + b2r
  k_wmma_gemm_bf16<4, 4><<<CDIV(tiles1, 8), B, 0, stream>>>(h_bf, w2l_pk, b2l, hl2, N);
  k_wmma_gemm_bf16<4, 4><<<CDIV(tiles1, 8), B, 0, stream>>>(h_bf, w2r_pk, b2r, hr2, N);

  // 5) layer-2 segment softmax + aggregation
  k_fill_f32<<<CDIV(N, B), B, 0, stream>>>(m2, -FLT_MAX, N);
  hipMemsetAsync(dn2, 0, (size_t)N * 4, stream);
  hipMemsetAsync(agg2, 0, (size_t)N * 64 * 4, stream);
  k_edge_logits_l2<<<CDIV(Etot * 32, B), B, 0, stream>>>(ei, E, Etot, hl2, hr2, att2, lg2, m2);
  k_edge_softmax_l2<<<CDIV(Etot, B), B, 0, stream>>>(ei, E, Etot, lg2, m2, dn2);
  k_edge_aggregate_l2<<<CDIV(Etot * 32, B), B, 0, stream>>>(ei, E, Etot, hl2, lg2, agg2);
  k_node_finalize_l2<<<CDIV(N * 32, B), B, 0, stream>>>(agg2, dn2, bias2, g2, be2, zbuf, N);

  // 6) decode
  k_decode<<<CDIV(EL * 32, B), B, 0, stream>>>(eli, EL, zbuf, out);
  (void)out_size;
}
